// LSTMWithProperAttention_60395830116684
// MI455X (gfx1250) — compile-verified, matrix operand order checked by reading, not run on hardware
//
#include <hip/hip_runtime.h>
#include <hip/hip_bf16.h>
#include <math.h>

typedef __attribute__((ext_vector_type(16))) _Float16 v16h;
typedef __attribute__((ext_vector_type(8)))  _Float16 v8h;
typedef __attribute__((ext_vector_type(4)))  _Float16 v4h;
typedef __attribute__((ext_vector_type(8)))  float    v8f;

#define Bn 256
#define Tn 512
#define Hn 128
#define Ln 128
#define INn 3
#define HS 160   // augmented LDS row: [h(128) | ext(32)]; 320B rows, 32B aligned
#define KW 160   // packed weight row stride (f16)

// ---- workspace layout (bytes) ----
static constexpr size_t kW0cat  = 0;                                    // [Whh0 | Wih0 | b0 | 0] 512x160
static constexpr size_t kWih1   = kW0cat  + (size_t)512 * KW * 2;       // plain Wih1 512x128
static constexpr size_t kW1hcat = kWih1   + (size_t)512 * 128 * 2;      // [Whh1 | b1 | 0] 512x160
static constexpr size_t kWacat  = kW1hcat + (size_t)512 * KW * 2;       // [Wa | bWa | 0] 128x160
static constexpr size_t kUacat  = kWacat  + (size_t)128 * KW * 2;       // [Ua | bUa | 0] 128x160
static constexpr size_t kH1     = kUacat  + (size_t)128 * KW * 2;
static constexpr size_t kC1     = kH1     + (size_t)256 * 128 * 4;
static constexpr size_t kH2     = kC1     + (size_t)256 * 128 * 4;
static constexpr size_t kC2     = kH2     + (size_t)256 * 128 * 4;
static constexpr size_t kLo     = (size_t)2 << 20;                      // lstm_out f16 (32MB, L2-resident)
static constexpr size_t kLoWa   = kLo + (size_t)Bn * Tn * Hn * 2;       // lo @ Wa^T + bWa, f16 (32MB)

// fast device math: v_exp_f32 / v_rcp_f32 based
__device__ inline float sigf(float x) { return __builtin_amdgcn_rcpf(1.0f + __expf(-x)); }
__device__ inline float ftanh(float x) {
  float e = __expf(2.0f * x);
  return 1.0f - 2.0f * __builtin_amdgcn_rcpf(e + 1.0f);
}

__device__ inline v8f wmma_f16(v16h a, v16h b, v8f c) {
  return __builtin_amdgcn_wmma_f32_16x16x32_f16(false, a, false, b, (short)0, c, false, false);
}

// ---------------- weight packing kernels ----------------
__global__ void cvt_f16_kernel(const float* __restrict__ src, _Float16* __restrict__ dst, int n) {
  int i = blockIdx.x * blockDim.x + threadIdx.x;
  if (i < n) dst[i] = (_Float16)src[i];
}

// rows: [Whh0[n][0..127] | Wih0[n][0..2] | b_ih0[n]+b_hh0[n] | 0...]
__global__ void pack_w0_kernel(const float* __restrict__ Whh, const float* __restrict__ Wih,
                               const float* __restrict__ bi, const float* __restrict__ bh,
                               _Float16* __restrict__ dst) {
  int n = blockIdx.x, k = threadIdx.x;
  float v;
  if (k < 128)      v = Whh[n * 128 + k];
  else if (k < 131) v = Wih[n * 3 + (k - 128)];
  else if (k == 131) v = bi[n] + bh[n];
  else              v = 0.0f;
  dst[(size_t)n * KW + k] = (_Float16)v;
}

// rows: [W[n][0..127] | bias[n] | 0...]  (bias = bi+bh, or single b when bh==nullptr)
__global__ void pack_wb_kernel(const float* __restrict__ W, const float* __restrict__ bi,
                               const float* __restrict__ bh, _Float16* __restrict__ dst) {
  int n = blockIdx.x, k = threadIdx.x;
  float v;
  if (k < 128)       v = W[n * 128 + k];
  else if (k == 128) v = bh ? (bi[n] + bh[n]) : bi[n];
  else               v = 0.0f;
  dst[(size_t)n * KW + k] = (_Float16)v;
}

// ---------------- encoder: 2-layer LSTM, 16 batch rows / WG ----------------
__global__ __launch_bounds__(256)
__attribute__((amdgpu_num_vgpr(256)))
void enc_kernel(const float* __restrict__ x, const int* __restrict__ lengths,
                const _Float16* __restrict__ W0cat, const _Float16* __restrict__ Wih1h,
                const _Float16* __restrict__ W1hcat,
                _Float16* __restrict__ lo16,
                float* __restrict__ h1s, float* __restrict__ c1s,
                float* __restrict__ h2s, float* __restrict__ c2s) {
  __shared__ _Float16 sh1[16 * HS];  // [h1 | x_t(3) | 1 | 0...]
  __shared__ _Float16 sh2[16 * HS];  // [h2 | 1 | 0...]

  const int tid = threadIdx.x;
  const int wave = tid >> 5, lane = tid & 31;
  const int col = lane & 15, half = lane >> 4;
  const int u0 = wave * 16, hcol = u0 + col;
  const int bb = blockIdx.x * 16;

  for (int i = tid; i < 16 * HS; i += 256) { sh1[i] = (_Float16)0.0f; sh2[i] = (_Float16)0.0f; }
  __syncthreads();
  if (tid < 16) {
    sh1[tid * HS + 131] = (_Float16)1.0f;  // layer0 bias column
    sh2[tid * HS + 128] = (_Float16)1.0f;  // layer1/attention bias column
  }

  // warm packed weights into cache hierarchy (global_prefetch_b8)
  for (int i = tid * 64; i < 512 * KW; i += 256 * 64) {
    __builtin_prefetch(W0cat + i, 0, 3);
    __builtin_prefetch(W1hcat + i, 0, 3);
  }
  for (int i = tid * 64; i < 512 * 128; i += 256 * 64) __builtin_prefetch(Wih1h + i, 0, 3);

  int lenr[8];
#pragma unroll
  for (int r = 0; r < 8; ++r) lenr[r] = lengths[bb + r + 8 * half];

  v8f c1, c2;
#pragma unroll
  for (int r = 0; r < 8; ++r) { c1[r] = 0.0f; c2[r] = 0.0f; }
  __syncthreads();

  for (int t = 0; t < Tn; ++t) {
    // opaque pointer laundering: stop LICM from hoisting the (un-registerable)
    // loop-invariant weight tiles -> keep them as per-step global_load_b128 from L2
    const _Float16* W0p  = W0cat;
    const _Float16* W1ip = Wih1h;
    const _Float16* W1hp = W1hcat;
    asm volatile("" : "+s"(W0p), "+s"(W1ip), "+s"(W1hp));

    // stage x_t into sh1 extension columns (128..130)
    if (tid < 48) {
      int bl = tid / 3, k = tid % 3;
      sh1[bl * HS + 128 + k] = (_Float16)x[((size_t)(bb + bl) * Tn + t) * INn + k];
    }
    __syncthreads();

    // ---- layer 0: gates = [h1|x|1] @ [Whh0|Wih0|b0]^T  (K=160, zero-init acc) ----
    v8f acc[4];
#pragma unroll
    for (int g = 0; g < 4; ++g)
#pragma unroll
      for (int r = 0; r < 8; ++r) acc[g][r] = 0.0f;
#pragma unroll
    for (int kk = 0; kk < 5; ++kk) {
      int k0 = kk * 32 + half * 16;
      v16h a = *(const v16h*)&sh1[col * HS + k0];
#pragma unroll
      for (int g = 0; g < 4; ++g) {
        v16h b = *(const v16h*)&W0p[(size_t)(g * 128 + hcol) * KW + k0];
        acc[g] = wmma_f16(a, b, acc[g]);
      }
    }
    float h1n[8];
#pragma unroll
    for (int r = 0; r < 8; ++r) {
      int m = r + 8 * half;
      bool v = (t < lenr[r]);
      float ig = sigf(acc[0][r]), fg = sigf(acc[1][r]);
      float gg = ftanh(acc[2][r]), og = sigf(acc[3][r]);
      float cn = fg * c1[r] + ig * gg;
      float hn = og * ftanh(cn);
      float hold = (float)sh1[m * HS + hcol];
      c1[r] = v ? cn : c1[r];
      h1n[r] = v ? hn : hold;
    }
    __syncthreads();
#pragma unroll
    for (int r = 0; r < 8; ++r) sh1[(r + 8 * half) * HS + hcol] = (_Float16)h1n[r];
    __syncthreads();

    // ---- layer 1: gates = h1@Wih1^T (K=128) + [h2|1] @ [Whh1|b1]^T (K=160) ----
#pragma unroll
    for (int g = 0; g < 4; ++g)
#pragma unroll
      for (int r = 0; r < 8; ++r) acc[g][r] = 0.0f;
#pragma unroll
    for (int kk = 0; kk < 4; ++kk) {
      int k0 = kk * 32 + half * 16;
      v16h a = *(const v16h*)&sh1[col * HS + k0];
#pragma unroll
      for (int g = 0; g < 4; ++g) {
        v16h b = *(const v16h*)&W1ip[(size_t)(g * 128 + hcol) * 128 + k0];
        acc[g] = wmma_f16(a, b, acc[g]);
      }
    }
#pragma unroll
    for (int kk = 0; kk < 5; ++kk) {
      int k0 = kk * 32 + half * 16;
      v16h a = *(const v16h*)&sh2[col * HS + k0];
#pragma unroll
      for (int g = 0; g < 4; ++g) {
        v16h b = *(const v16h*)&W1hp[(size_t)(g * 128 + hcol) * KW + k0];
        acc[g] = wmma_f16(a, b, acc[g]);
      }
    }
    float h2n[8];
#pragma unroll
    for (int r = 0; r < 8; ++r) {
      int m = r + 8 * half;
      bool v = (t < lenr[r]);
      float ig = sigf(acc[0][r]), fg = sigf(acc[1][r]);
      float gg = ftanh(acc[2][r]), og = sigf(acc[3][r]);
      float cn = fg * c2[r] + ig * gg;
      float hn = og * ftanh(cn);
      float hold = (float)sh2[m * HS + hcol];
      c2[r] = v ? cn : c2[r];
      h2n[r] = v ? hn : hold;
      lo16[((size_t)(bb + m) * Tn + t) * Hn + hcol] = (_Float16)(v ? hn : 0.0f);
    }
    __syncthreads();
#pragma unroll
    for (int r = 0; r < 8; ++r) sh2[(r + 8 * half) * HS + hcol] = (_Float16)h2n[r];
    __syncthreads();
  }

#pragma unroll
  for (int r = 0; r < 8; ++r) {
    int m = r + 8 * half;
    int gi = (bb + m) * Hn + hcol;
    c1s[gi] = c1[r];
    c2s[gi] = c2[r];
    h1s[gi] = (float)sh1[m * HS + hcol];
    h2s[gi] = (float)sh2[m * HS + hcol];
  }
}

// ---------------- loWa = lo @ Wa^T + bWa  (M=B*T, N=K=128) ----------------
__global__ __launch_bounds__(256)
void wa_gemm_kernel(const _Float16* __restrict__ lo16, const _Float16* __restrict__ Wacat,
                    const float* __restrict__ bWa, _Float16* __restrict__ loWah) {
  const int tid = threadIdx.x, wave = tid >> 5, lane = tid & 31;
  const int col = lane & 15, half = lane >> 4;
  const size_t rowbase = (size_t)blockIdx.x * 16;
  const int n = wave * 16 + col;

  v8f acc;
  float bv = bWa[n];
#pragma unroll
  for (int r = 0; r < 8; ++r) acc[r] = bv;
#pragma unroll
  for (int kk = 0; kk < 4; ++kk) {
    int k0 = kk * 32 + half * 16;
    v16h a = *(const v16h*)&lo16[(rowbase + col) * Hn + k0];
    v16h b = *(const v16h*)&Wacat[(size_t)n * KW + k0];
    acc = wmma_f16(a, b, acc);
  }
#pragma unroll
  for (int r = 0; r < 8; ++r)
    loWah[(rowbase + r + 8 * half) * Hn + n] = (_Float16)acc[r];
}

// ---------------- fused decoder: 512 threads; waves 0-7 = WMMA LSTM, all = attention ----------------
__global__ __launch_bounds__(512)
__attribute__((amdgpu_num_vgpr(256)))
void dec_kernel(const float* __restrict__ y, const int* __restrict__ lengths,
                const _Float16* __restrict__ W0cat, const _Float16* __restrict__ Wih1h,
                const _Float16* __restrict__ W1hcat,
                const _Float16* __restrict__ Wacat, const _Float16* __restrict__ Uacat,
                const float* __restrict__ Va, const float* __restrict__ bVa,
                const float* __restrict__ fc1w, const float* __restrict__ fc1b,
                const float* __restrict__ fc2w, const float* __restrict__ fc2b,
                _Float16* __restrict__ lo16, _Float16* __restrict__ loWah,
                const float* __restrict__ h1s, const float* __restrict__ c1s,
                const float* __restrict__ h2s, const float* __restrict__ c2s,
                float* __restrict__ out) {
  __shared__ _Float16 sh1[16 * HS];  // [h1 | y_t(3) | 1 | 0...]
  __shared__ _Float16 sh2[16 * HS];  // [h2 | 1 | 0...]
  __shared__ float sx[16 * 4];       // async-staged y_t (f32)
  __shared__ float sUa[16 * Hn];     // hUa + bUa, f32
  __shared__ float sCtx[16 * Hn];    // attention context
  __shared__ float sScore[16 * Tn];  // scores -> softmax weights
  __shared__ float sR1[16 * 32];     // fc1 activations
  __shared__ float sRed[16 * 32];
  __shared__ float sMax[16];
  __shared__ float sInv[16];
  __shared__ float sVa[Hn];

  const int tid = threadIdx.x, wave = tid >> 5, lane = tid & 31;
  const int col = lane & 15, half = lane >> 4;
  const int u0 = wave * 16, hcol = u0 + col;  // valid for wave < 8
  const int bb = blockIdx.x * 16;

  // zero augmented h buffers, then stage state + constants
  for (int i = tid; i < 16 * HS; i += 512) { sh1[i] = (_Float16)0.0f; sh2[i] = (_Float16)0.0f; }
  __syncthreads();
  for (int i = tid; i < 16 * Hn; i += 512) {
    int m = i >> 7, h = i & 127;
    sh1[m * HS + h] = (_Float16)h1s[(bb + m) * Hn + h];
    sh2[m * HS + h] = (_Float16)h2s[(bb + m) * Hn + h];
  }
  if (tid < 16) {
    sh1[tid * HS + 131] = (_Float16)1.0f;
    sh2[tid * HS + 128] = (_Float16)1.0f;
  }
  if (tid < Hn) sVa[tid] = Va[tid];

  v8f c1, c2;
  if (wave < 8) {
#pragma unroll
    for (int r = 0; r < 8; ++r) {
      int m = r + 8 * half;
      int gi = (bb + m) * Hn + hcol;
      c1[r] = c1s[gi];
      c2[r] = c2s[gi];
    }
  }
  const float bva = bVa[0];
  __syncthreads();

  for (int ii = 0; ii < Ln; ++ii) {
    // opaque pointer laundering: keep weight tiles as per-step L2 streams,
    // not register/scratch-resident invariants
    const _Float16* W0p  = W0cat;
    const _Float16* W1ip = Wih1h;
    const _Float16* W1hp = W1hcat;
    const _Float16* Wap  = Wacat;
    const _Float16* Uap  = Uacat;
    asm volatile("" : "+s"(W0p), "+s"(W1ip), "+s"(W1hp), "+s"(Wap), "+s"(Uap));

    if (ii > 0) {
      // ---- stage teacher input y[:, ii-1] via async global->LDS, then cvt to f16 ext cols ----
      if (tid < 48) {
        int bl = tid / 3, k = tid % 3;
        const float* gp = &y[((size_t)(bb + bl) * Ln + (ii - 1)) * INn + k];
        unsigned ldsoff = (unsigned)(unsigned long long)(void*)&sx[bl * 4 + k];
        asm volatile("global_load_async_to_lds_b32 %0, %1, off"
                     :: "v"(ldsoff), "v"(gp) : "memory");
      }
      asm volatile("s_wait_asynccnt 0x0" ::: "memory");
      if (tid < 48) {
        int bl = tid / 3, k = tid % 3;
        sh1[bl * HS + 128 + k] = (_Float16)sx[bl * 4 + k];
      }
      __syncthreads();

      v8f acc[4];
      float h1n[8], h2n[8];
      if (wave < 8) {
#pragma unroll
        for (int g = 0; g < 4; ++g)
#pragma unroll
          for (int r = 0; r < 8; ++r) acc[g][r] = 0.0f;
#pragma unroll
        for (int kk = 0; kk < 5; ++kk) {
          int k0 = kk * 32 + half * 16;
          v16h a = *(const v16h*)&sh1[col * HS + k0];
#pragma unroll
          for (int g = 0; g < 4; ++g) {
            v16h b = *(const v16h*)&W0p[(size_t)(g * 128 + hcol) * KW + k0];
            acc[g] = wmma_f16(a, b, acc[g]);
          }
        }
#pragma unroll
        for (int r = 0; r < 8; ++r) {
          float ig = sigf(acc[0][r]), fg = sigf(acc[1][r]);
          float gg = ftanh(acc[2][r]), og = sigf(acc[3][r]);
          c1[r] = fg * c1[r] + ig * gg;
          h1n[r] = og * ftanh(c1[r]);
        }
      }
      __syncthreads();
      if (wave < 8) {
#pragma unroll
        for (int r = 0; r < 8; ++r) sh1[(r + 8 * half) * HS + hcol] = (_Float16)h1n[r];
      }
      __syncthreads();
      if (wave < 8) {
#pragma unroll
        for (int g = 0; g < 4; ++g)
#pragma unroll
          for (int r = 0; r < 8; ++r) acc[g][r] = 0.0f;
#pragma unroll
        for (int kk = 0; kk < 4; ++kk) {
          int k0 = kk * 32 + half * 16;
          v16h a = *(const v16h*)&sh1[col * HS + k0];
#pragma unroll
          for (int g = 0; g < 4; ++g) {
            v16h b = *(const v16h*)&W1ip[(size_t)(g * 128 + hcol) * 128 + k0];
            acc[g] = wmma_f16(a, b, acc[g]);
          }
        }
#pragma unroll
        for (int kk = 0; kk < 5; ++kk) {
          int k0 = kk * 32 + half * 16;
          v16h a = *(const v16h*)&sh2[col * HS + k0];
#pragma unroll
          for (int g = 0; g < 4; ++g) {
            v16h b = *(const v16h*)&W1hp[(size_t)(g * 128 + hcol) * KW + k0];
            acc[g] = wmma_f16(a, b, acc[g]);
          }
        }
#pragma unroll
        for (int r = 0; r < 8; ++r) {
          int m = r + 8 * half;
          float ig = sigf(acc[0][r]), fg = sigf(acc[1][r]);
          float gg = ftanh(acc[2][r]), og = sigf(acc[3][r]);
          c2[r] = fg * c2[r] + ig * gg;
          h2n[r] = og * ftanh(c2[r]);
          lo16[((size_t)(bb + m) * Tn + ii) * Hn + hcol] = (_Float16)h2n[r];  // lo[:, ii] = h2
        }
      }
      __syncthreads();
      if (wave < 8) {
#pragma unroll
        for (int r = 0; r < 8; ++r) sh2[(r + 8 * half) * HS + hcol] = (_Float16)h2n[r];
      }
      __syncthreads();
      if (wave < 8) {
        // ---- incremental loWa row: loWa[:, ii] = [h2|1] @ [Wa|bWa]^T ----
        v8f wacc;
#pragma unroll
        for (int r = 0; r < 8; ++r) wacc[r] = 0.0f;
#pragma unroll
        for (int kk = 0; kk < 5; ++kk) {
          int k0 = kk * 32 + half * 16;
          v16h a = *(const v16h*)&sh2[col * HS + k0];
          v16h b = *(const v16h*)&Wap[(size_t)hcol * KW + k0];
          wacc = wmma_f16(a, b, wacc);
        }
#pragma unroll
        for (int r = 0; r < 8; ++r)
          loWah[((size_t)(bb + r + 8 * half) * Tn + ii) * Hn + hcol] = (_Float16)wacc[r];
      }
    }

    // ---- sUa = [h2|1] @ [Ua|bUa]^T (every step, also feeds out0 at ii==0) ----
    if (wave < 8) {
      v8f uacc;
#pragma unroll
      for (int r = 0; r < 8; ++r) uacc[r] = 0.0f;
#pragma unroll
      for (int kk = 0; kk < 5; ++kk) {
        int k0 = kk * 32 + half * 16;
        v16h a = *(const v16h*)&sh2[col * HS + k0];
        v16h b = *(const v16h*)&Uap[(size_t)hcol * KW + k0];
        uacc = wmma_f16(a, b, uacc);
      }
#pragma unroll
      for (int r = 0; r < 8; ++r) sUa[(r + 8 * half) * Hn + hcol] = uacc[r];
    }
    __threadfence();  // make lo16/loWah row writes visible to in-WG readers
    __syncthreads();

    // ---- scores: tanh(loWa[t] + hUa) . Va + bVa, masked (all 16 waves) ----
    {
      int bl = tid >> 5, ch = tid & 31;
      int len = lengths[bb + bl];
      const float* ua = &sUa[bl * Hn];
      float mx = -1e30f;
      for (int tt = ch * 16; tt < ch * 16 + 16; ++tt) {
        bool valid = (tt < len) || (tt <= ii);
        float sc = -1e30f;
        if (valid) {
          const v8h* lp8 = (const v8h*)&loWah[((size_t)(bb + bl) * Tn + tt) * Hn];
          float a = bva;
          for (int hh = 0; hh < Hn / 8; ++hh) {
            v8h lv = lp8[hh];
#pragma unroll
            for (int q = 0; q < 8; ++q)
              a += sVa[hh * 8 + q] * ftanh((float)lv[q] + ua[hh * 8 + q]);
          }
          sc = a;
        }
        sScore[bl * Tn + tt] = sc;
        mx = fmaxf(mx, sc);
      }
      sRed[bl * 32 + ch] = mx;
    }
    __syncthreads();
    if (tid < 16) {
      float m = sRed[tid * 32];
      for (int q = 1; q < 32; ++q) m = fmaxf(m, sRed[tid * 32 + q]);
      sMax[tid] = m;
    }
    __syncthreads();
    {
      int bl = tid >> 5, ch = tid & 31;
      float bm = sMax[bl], ps = 0.0f;
      for (int tt = ch * 16; tt < ch * 16 + 16; ++tt) {
        float e = __expf(sScore[bl * Tn + tt] - bm);
        sScore[bl * Tn + tt] = e;
        ps += e;
      }
      sRed[bl * 32 + ch] = ps;
    }
    __syncthreads();
    if (tid < 16) {
      float s = 0.0f;
      for (int q = 0; q < 32; ++q) s += sRed[tid * 32 + q];
      sInv[tid] = __builtin_amdgcn_rcpf(s);
    }
    __syncthreads();

    // ---- context: ctx = softmax(scores) . lo  (v4h loads) ----
    {
      int bl = tid >> 5, pr = tid & 31;
      int h0 = pr * 4;
      float a0 = 0.0f, a1 = 0.0f, a2 = 0.0f, a3 = 0.0f;
      const _Float16* lp = &lo16[(size_t)(bb + bl) * Tn * Hn + h0];
      for (int tt = 0; tt < Tn; ++tt) {
        float w = sScore[bl * Tn + tt];
        v4h lv = *(const v4h*)(lp + (size_t)tt * Hn);
        a0 += w * (float)lv[0];
        a1 += w * (float)lv[1];
        a2 += w * (float)lv[2];
        a3 += w * (float)lv[3];
      }
      float inv = sInv[bl];
      sCtx[bl * Hn + h0 + 0] = a0 * inv;
      sCtx[bl * Hn + h0 + 1] = a1 * inv;
      sCtx[bl * Hn + h0 + 2] = a2 * inv;
      sCtx[bl * Hn + h0 + 3] = a3 * inv;
    }
    __syncthreads();

    // ---- head: fc2(relu(fc1(ctx))) ----
    {
      int bl = tid >> 5, u = tid & 31;
      float a = fc1b[u];
      for (int h = 0; h < Hn; ++h) a += sCtx[bl * Hn + h] * fc1w[u * Hn + h];
      sR1[bl * 32 + u] = fmaxf(a, 0.0f);
    }
    __syncthreads();
    if (tid < 32) {
      int bl = tid >> 1, o = tid & 1;
      float a = fc2b[o];
      for (int u = 0; u < 32; ++u) a += sR1[bl * 32 + u] * fc2w[o * 32 + u];
      out[((size_t)(bb + bl) * Ln + ii) * 2 + o] = a;
    }
    __syncthreads();
  }
}

extern "C" void kernel_launch(void* const* d_in, const int* in_sizes, int n_in,
                              void* d_out, int out_size, void* d_ws, size_t ws_size,
                              hipStream_t stream) {
  const float* x     = (const float*)d_in[0];
  const float* y     = (const float*)d_in[1];
  const int* lengths = (const int*)d_in[3];
  const float* Wih0  = (const float*)d_in[4];
  const float* Whh0  = (const float*)d_in[5];
  const float* bih0  = (const float*)d_in[6];
  const float* bhh0  = (const float*)d_in[7];
  const float* Wih1  = (const float*)d_in[8];
  const float* Whh1  = (const float*)d_in[9];
  const float* bih1  = (const float*)d_in[10];
  const float* bhh1  = (const float*)d_in[11];
  const float* Wa    = (const float*)d_in[12];
  const float* bWa   = (const float*)d_in[13];
  const float* Ua    = (const float*)d_in[14];
  const float* bUa   = (const float*)d_in[15];
  const float* Va    = (const float*)d_in[16];
  const float* bVa   = (const float*)d_in[17];
  const float* fc1w  = (const float*)d_in[18];
  const float* fc1b  = (const float*)d_in[19];
  const float* fc2w  = (const float*)d_in[20];
  const float* fc2b  = (const float*)d_in[21];

  char* ws = (char*)d_ws;
  _Float16* W0cat  = (_Float16*)(ws + kW0cat);
  _Float16* Wih1h  = (_Float16*)(ws + kWih1);
  _Float16* W1hcat = (_Float16*)(ws + kW1hcat);
  _Float16* Wacat  = (_Float16*)(ws + kWacat);
  _Float16* Uacat  = (_Float16*)(ws + kUacat);
  float* h1s = (float*)(ws + kH1);
  float* c1s = (float*)(ws + kC1);
  float* h2s = (float*)(ws + kH2);
  float* c2s = (float*)(ws + kC2);
  _Float16* lo16  = (_Float16*)(ws + kLo);
  _Float16* loWah = (_Float16*)(ws + kLoWa);

  // pack weights (fold input weights + biases into augmented-K f16 rows)
  pack_w0_kernel<<<512, KW, 0, stream>>>(Whh0, Wih0, bih0, bhh0, W0cat);
  cvt_f16_kernel<<<(512 * 128 + 255) / 256, 256, 0, stream>>>(Wih1, Wih1h, 512 * 128);
  pack_wb_kernel<<<512, KW, 0, stream>>>(Whh1, bih1, bhh1, W1hcat);
  pack_wb_kernel<<<128, KW, 0, stream>>>(Wa, bWa, nullptr, Wacat);
  pack_wb_kernel<<<128, KW, 0, stream>>>(Ua, bUa, nullptr, Uacat);

  // encoder: 16 WGs x 8 waves, WMMA recurrent LSTM
  enc_kernel<<<16, 256, 0, stream>>>(x, lengths, W0cat, Wih1h, W1hcat,
                                     lo16, h1s, c1s, h2s, c2s);

  // loWa = lo @ Wa^T + bWa over all B*T rows (bulk WMMA GEMM)
  wa_gemm_kernel<<<(Bn * Tn) / 16, 256, 0, stream>>>(lo16, Wacat, bWa, loWah);

  // fused decoder: 512 threads (waves 0-7 WMMA, all waves attention)
  dec_kernel<<<16, 512, 0, stream>>>(y, lengths, W0cat, Wih1h, W1hcat, Wacat, Uacat,
                                     Va, bVa, fc1w, fc1b, fc2w, fc2b, lo16, loWah,
                                     h1s, c1s, h2s, c2s, (float*)d_out);
}
